// GLMRNN_59811714564461
// MI455X (gfx1250) — matrix operand order repaired and checked by reading, not compiled, not run on hardware
//
#include <hip/hip_runtime.h>
#include <cstdint>
#include <cstddef>

typedef __attribute__((ext_vector_type(16))) _Float16 v16h;
typedef __attribute__((ext_vector_type(8)))  float    v8f;

#define N_NEUR   1024
#define NWG      64
#define ROWS_WG  16
#define DT_STEP  0.1f

union AB { uint4 q[2]; v16h h; };

// ---- stateless counter-based RNG: deterministic across graph replays ----
__device__ __forceinline__ float u01(unsigned a, unsigned b, unsigned c) {
  unsigned x = a * 0x9E3779B1u ^ (b + 0x85EBCA77u) ^ (c * 0xC2B2AE3Du + 0x27D4EB2Fu);
  x ^= x >> 16; x *= 0x7FEB352Du;
  x ^= x >> 15; x *= 0x846CA68Bu;
  x ^= x >> 16;
  return ((x >> 8) + 0.5f) * (1.0f / 16777216.0f);
}

__device__ __forceinline__ float softplus_f(float v) {
  return v > 20.0f ? v : log1pf(__expf(v));
}

// Knuth's Poisson sampler; lambda = g*dt is O(0.1), so ~1 iteration expected
__device__ __forceinline__ float poisson_knuth(float lam, unsigned t, unsigned n) {
  float L = __expf(-lam);
  float p = 1.0f;
  unsigned k = 0;
  do { p *= u01(t, n, k); ++k; } while (p > L && k < 64u);
  return (float)(k - 1);
}

// ---- sense(generation)-based grid barrier, device-scope atomics ----
__device__ __forceinline__ void grid_barrier(unsigned* cnt, unsigned* gen,
                                             unsigned nwg, int lane) {
  __threadfence();                                   // release our x publish
  if (lane == 0) {
    unsigned g = __hip_atomic_load(gen, __ATOMIC_RELAXED, __HIP_MEMORY_SCOPE_AGENT);
    unsigned arrived =
        __hip_atomic_fetch_add(cnt, 1u, __ATOMIC_ACQ_REL, __HIP_MEMORY_SCOPE_AGENT);
    if (arrived == nwg - 1u) {
      __hip_atomic_store(cnt, 0u, __ATOMIC_RELAXED, __HIP_MEMORY_SCOPE_AGENT);
      __hip_atomic_fetch_add(gen, 1u, __ATOMIC_RELEASE, __HIP_MEMORY_SCOPE_AGENT);
    } else {
      while (__hip_atomic_load(gen, __ATOMIC_ACQUIRE, __HIP_MEMORY_SCOPE_AGENT) == g)
        __builtin_amdgcn_s_sleep(2);
    }
  }
  __threadfence();                                   // acquire others' publishes
}

// Re-zero barrier words every launch; write the all-ones t=0 columns.
__global__ void glmrnn_init_kernel(float* __restrict__ out,
                                   const int* __restrict__ Tp,
                                   unsigned* __restrict__ bar) {
  const int i = blockIdx.x * blockDim.x + threadIdx.x;
  if (i < 2) bar[i] = 0u;
  const int T = Tp[0];
  const size_t NT = (size_t)N_NEUR * (size_t)T;
  if (i < N_NEUR) {
    out[(size_t)i * T]          = 1.0f;   // st[:,0]
    out[NT + (size_t)i * T]     = 1.0f;   // gt[:,0]
    out[2 * NT + (size_t)i * T] = 1.0f;   // xt[:,0]
  }
}

// Persistent kernel: 64 WGs x wave32. Each wave owns 16 rows of W, held as
// ready-to-issue WMMA A-operands in 256 VGPRs (gfx1250 1024-VGPR file) for
// the entire 4095-step recurrence. Per step: publish x2, grid barrier,
// 2KB x broadcast into LDS, 32 chained v_wmma_f32_16x16x32_f16, pointwise.
__global__ void __launch_bounds__(32)
glmrnn_step_kernel(const float* __restrict__ W,
                   const float* __restrict__ Bvec,
                   const float* __restrict__ tau,
                   const int*   __restrict__ Tp,
                   float* __restrict__ out,
                   unsigned* __restrict__ bar,
                   _Float16* __restrict__ xbuf) {
  __shared__ _Float16 xlds[N_NEUR];            // 2 KB: current x, broadcast source

  const int lane = threadIdx.x;
  const int wg   = blockIdx.x;
  const int r0   = wg * ROWS_WG;
  const int T    = Tp[0];
  const size_t NT = (size_t)N_NEUR * (size_t)T;

  const int koffA = (lane >> 4) * 8;    // A: lower half-wave K 0..7/16..23, upper 8..15/24..31
  const int koffB = (lane >> 4) * 8;    // B: lower half-wave K 0..7/16..23, upper 8..15/24..31

  // ---- Phase A: load this lane's A-operand slice of W straight into VGPRs ----
  // Lane (l&15) carries row r0+(l&15); per K-block it needs halfs
  // [kb*32+koffA .. +7] and [.. +16 .. +23] -> 4 float4 global loads, cvt to f16.
  const float* wrow = W + (size_t)(r0 + (lane & 15)) * N_NEUR;
  __builtin_prefetch(wrow, 0, 1);                       // global_prefetch_b8
  AB areg[32];
#pragma unroll
  for (int kb = 0; kb < 32; ++kb) {
    const int ka = kb * 32 + koffA;
    float4 f0 = *(const float4*)(wrow + ka);
    float4 f1 = *(const float4*)(wrow + ka + 4);
    float4 f2 = *(const float4*)(wrow + ka + 16);
    float4 f3 = *(const float4*)(wrow + ka + 20);
    v16h h;
    h[0]  = (_Float16)f0.x; h[1]  = (_Float16)f0.y; h[2]  = (_Float16)f0.z; h[3]  = (_Float16)f0.w;
    h[4]  = (_Float16)f1.x; h[5]  = (_Float16)f1.y; h[6]  = (_Float16)f1.z; h[7]  = (_Float16)f1.w;
    h[8]  = (_Float16)f2.x; h[9]  = (_Float16)f2.y; h[10] = (_Float16)f2.z; h[11] = (_Float16)f2.w;
    h[12] = (_Float16)f3.x; h[13] = (_Float16)f3.y; h[14] = (_Float16)f3.z; h[15] = (_Float16)f3.w;
    areg[kb].h = h;
  }

  // Owner lanes: lane 0 -> rows r0..r0+7, lane 16 -> rows r0+8..r0+15
  // (WMMA f32 16x16 C/D layout: VGPR m = row m, +8 for the upper half-wave)
  const bool owner = ((lane & 15) == 0);
  const int  rbase = r0 + (lane >> 4) * 8;
  const float decay = 1.0f - DT_STEP / tau[0];
  float xs[8], ss[8], Bl[8];
#pragma unroll
  for (int j = 0; j < 8; ++j) {
    xs[j] = 1.0f; ss[j] = 1.0f;
    Bl[j] = owner ? Bvec[rbase + j] : 0.0f;
  }

  for (int t = 1; t < T; ++t) {
    _Float16* xpub = xbuf + (size_t)(t & 1) * N_NEUR;   // parity double buffer

    // x2 = decay*x + s for owned rows; emit xt column t; publish f16 x2
    if (owner) {
#pragma unroll
      for (int j = 0; j < 8; ++j) {
        float x2 = decay * xs[j] + ss[j];
        xs[j] = x2;
        out[2 * NT + (size_t)(rbase + j) * T + t] = x2;
        xpub[rbase + j] = (_Float16)x2;
      }
    }

    grid_barrier(&bar[0], &bar[1], NWG, lane);

    // Pull the freshly published x (2 KB f16) into LDS with agent-scope loads
    {
      const unsigned long long* xg = (const unsigned long long*)xpub;
      unsigned long long* xl = (unsigned long long*)xlds;
#pragma unroll
      for (int i = 0; i < 8; ++i)
        xl[lane * 8 + i] = __hip_atomic_load(&xg[lane * 8 + i],
                                             __ATOMIC_RELAXED, __HIP_MEMORY_SCOPE_AGENT);
    }
    __syncthreads();

    // ---- y[16 rows] = W_slice @ x : 32 chained WMMAs, A straight from VGPRs,
    // B = x chunk replicated across all 16 columns (2 broadcast LDS loads each)
    v8f acc = {};
#pragma unroll
    for (int kb = 0; kb < 32; ++kb) {
      AB b;
      const int kx = kb * 32 + koffB;
      b.q[0] = *(const uint4*)&xlds[kx];               // K +0..7
      b.q[1] = *(const uint4*)&xlds[kx + 16];          // K +16..23
      acc = __builtin_amdgcn_wmma_f32_16x16x32_f16(false, areg[kb].h, false, b.h,
                                                   (short)0, acc, false, false);
    }

    // softplus + Poisson + emit gt/st column t, update recurrent state
    if (owner) {
#pragma unroll
      for (int j = 0; j < 8; ++j) {
        float g = softplus_f(acc[j] + Bl[j]);
        out[NT + (size_t)(rbase + j) * T + t] = g;
        float s = poisson_knuth(DT_STEP * g, (unsigned)t, (unsigned)(rbase + j));
        out[(size_t)(rbase + j) * T + t] = s;
        ss[j] = s;
      }
    }
    __syncthreads();
  }
}

extern "C" void kernel_launch(void* const* d_in, const int* in_sizes, int n_in,
                              void* d_out, int out_size, void* d_ws, size_t ws_size,
                              hipStream_t stream) {
  const float* W   = (const float*)d_in[0];
  const float* Bv  = (const float*)d_in[1];
  const float* tau = (const float*)d_in[2];
  const int*   Tp  = (const int*)d_in[3];
  float* out = (float*)d_out;

  unsigned* bar  = (unsigned*)d_ws;                    // 2 words
  _Float16* xbuf = (_Float16*)((char*)d_ws + 256);     // 2 x 1024 f16

  glmrnn_init_kernel<<<dim3(8), dim3(128), 0, stream>>>(out, Tp, bar);
  glmrnn_step_kernel<<<dim3(NWG), dim3(32), 0, stream>>>(W, Bv, tau, Tp, out, bar, xbuf);

  (void)in_sizes; (void)n_in; (void)out_size; (void)ws_size;
}
